// TreeModel_17523466568298
// MI455X (gfx1250) — compile-verified, matrix operand order checked by reading, not run on hardware
//
#include <hip/hip_runtime.h>

#define B_ROWS 131072
#define D_DIM  256
#define TILE_M 128
#define PADH   272            // LDS row stride in halves (breaks bank conflicts)
#define KSTEPS 8              // 256 / 32
#define WAVES  8
#define BLK    (WAVES * 32)   // 256 threads, 8 wave32s

typedef __attribute__((ext_vector_type(16))) __bf16          v16bf;
typedef __attribute__((ext_vector_type(8)))  float           v8f;
typedef __attribute__((ext_vector_type(4)))  float           f4;
typedef __attribute__((ext_vector_type(4)))  unsigned short  u16x4;

union Frag { f4 q[2]; v16bf v; };

// ---- workspace layout (int units) ----
#define WS_COUNTS 0
#define WS_CURS   4
#define WS_OFFS   8                 // offs[0..4], offs[4] = padded total
#define WS_IDX    16
#define WS_PERM   (16 + B_ROWS)
#define WS_PERM_LEN (B_ROWS + 4 * TILE_M)
#define WPACK_BYTES ((size_t)(WS_PERM + WS_PERM_LEN) * 4)   // bf16 packed weights after ints
#define WPLANE 65536                // ushorts per 256x256 plane

__device__ __forceinline__ unsigned short f2bf(float f) {
  unsigned u = __float_as_uint(f);
  u += 0x7FFFu + ((u >> 16) & 1u);          // round-to-nearest-even
  return (unsigned short)(u >> 16);
}
__device__ __forceinline__ float bf2f(unsigned short h) {
  return __uint_as_float(((unsigned)h) << 16);
}

// ---- K0: split fp32 weights into bf16 hi/lo planes, pre-swizzled to WMMA B-frag order ----
// B frag (32x16 bf16): lane (r = lane&15, hi = lane>>4) holds col n = nt*16+r;
// VGPR v<4: K = ks*32 + 8*hi + 2*v + h ; v>=4: K = ks*32 + 16 + 8*hi + 2*(v-4) + h
__global__ void k_pack(const float* __restrict__ Wl, const float* __restrict__ Wm,
                       const float* __restrict__ Wr, unsigned short* __restrict__ wp) {
  int gid = blockIdx.x * 256 + threadIdx.x;
  if (gid >= (7 << 15)) return;
  int v    = gid & 7;
  int lane = (gid >> 3) & 31;
  int ks   = (gid >> 8) & 7;
  int nt   = (gid >> 11) & 15;
  int mat  = gid >> 15;
  const float* W = (mat < 4) ? (Wl + mat * 65536)
                             : ((mat < 6) ? (Wm + (mat - 4) * 65536) : Wr);
  int hi = lane >> 4, r = lane & 15;
  int n = nt * 16 + r;
  unsigned hw = 0, lw = 0;
  for (int h = 0; h < 2; ++h) {
    int k = ks * 32 + 8 * hi + ((v < 4) ? (2 * v + h) : (16 + 2 * (v - 4) + h));
    float w = W[k * D_DIM + n];
    unsigned short wh = f2bf(w);
    unsigned short wl = f2bf(w - bf2f(wh));
    hw |= ((unsigned)wh) << (16 * h);
    lw |= ((unsigned)wl) << (16 * h);
  }
  int idx = (nt * 8 + ks) * 512 + lane * 16 + v * 2;
  *(unsigned*)(wp + (mat * 2 + 0) * WPLANE + idx) = hw;
  *(unsigned*)(wp + (mat * 2 + 1) * WPLANE + idx) = lw;
}

// ---- K1: gate (argmax, first-max tiebreak) + expert counts ----
__global__ void k_gate(const float* __restrict__ x, const float* __restrict__ Wg,
                       const float* __restrict__ bg, int* __restrict__ wsi) {
  int row = blockIdx.x * 256 + threadIdx.x;
  if (row >= B_ROWS) return;
  f4 acc = {0.f, 0.f, 0.f, 0.f};
  const float* xr = x + (size_t)row * D_DIM;
#pragma unroll 4
  for (int d = 0; d < D_DIM; d += 4) {
    f4 xv = *(const f4*)(xr + d);
    acc += xv.x * *(const f4*)(Wg + (d + 0) * 4);
    acc += xv.y * *(const f4*)(Wg + (d + 1) * 4);
    acc += xv.z * *(const f4*)(Wg + (d + 2) * 4);
    acc += xv.w * *(const f4*)(Wg + (d + 3) * 4);
  }
  float l0 = acc.x + bg[0], l1 = acc.y + bg[1], l2 = acc.z + bg[2], l3 = acc.w + bg[3];
  int bi = 0; float bv = l0;
  if (l1 > bv) { bv = l1; bi = 1; }
  if (l2 > bv) { bv = l2; bi = 2; }
  if (l3 > bv) { bv = l3; bi = 3; }
  wsi[WS_IDX + row] = bi;
  atomicAdd(&wsi[WS_COUNTS + bi], 1);
}

// ---- K2: pad counts to TILE_M, exclusive scan, fill padding perm slots with -1 ----
__global__ void k_scan(int* __restrict__ wsi) {
  if (threadIdx.x != 0 || blockIdx.x != 0) return;
  int off = 0;
  for (int e = 0; e < 4; ++e) {
    int c = wsi[WS_COUNTS + e];
    int pad = (c + TILE_M - 1) / TILE_M * TILE_M;
    for (int p2 = off + c; p2 < off + pad; ++p2) wsi[WS_PERM + p2] = -1;
    wsi[WS_CURS + e] = off;
    wsi[WS_OFFS + e] = off;
    off += pad;
  }
  wsi[WS_OFFS + 4] = off;
}

// ---- K3: scatter rows into expert-grouped order ----
__global__ void k_scatter(int* __restrict__ wsi) {
  int row = blockIdx.x * 256 + threadIdx.x;
  if (row >= B_ROWS) return;
  int e = wsi[WS_IDX + row];
  int pos = atomicAdd(&wsi[WS_CURS + e], 1);
  wsi[WS_PERM + pos] = row;
}

// ---- bf16x3 GEMM layer: C = relu(in * W + bias); in/out in LDS hi/lo bf16 planes ----
template <bool LAST>
__device__ __forceinline__ void gemm_layer(
    const unsigned short* __restrict__ inH, const unsigned short* __restrict__ inL,
    const unsigned short* __restrict__ wH,  const unsigned short* __restrict__ wL,
    const float* __restrict__ bias,
    unsigned short* __restrict__ outH, unsigned short* __restrict__ outL,
    float* __restrict__ out, const int* __restrict__ rows_s) {
  const int lane = threadIdx.x & 31;
  const int wave = threadIdx.x >> 5;
  const int r = lane & 15, hi = lane >> 4, hi8 = hi * 8;
  const int nt0 = wave * 2;

  v8f acc[2][8];
#pragma unroll
  for (int t = 0; t < 2; ++t)
#pragma unroll
    for (int m = 0; m < 8; ++m)
#pragma unroll
      for (int q = 0; q < 8; ++q) acc[t][m][q] = 0.f;

  for (int ks = 0; ks < KSTEPS; ++ks) {
    Frag bH[2], bL[2];
#pragma unroll
    for (int t = 0; t < 2; ++t) {
      const unsigned short* ph = wH + ((nt0 + t) * 8 + ks) * 512 + lane * 16;
      const unsigned short* pl = wL + ((nt0 + t) * 8 + ks) * 512 + lane * 16;
      bH[t].q[0] = *(const f4*)(ph);     bH[t].q[1] = *(const f4*)(ph + 8);
      bL[t].q[0] = *(const f4*)(pl);     bL[t].q[1] = *(const f4*)(pl + 8);
    }
    if (ks + 1 < KSTEPS) {  // gfx1250 global_prefetch for next weight block
      __builtin_prefetch(wH + (nt0 * 8 + ks + 1) * 512 + lane * 16, 0, 1);
      __builtin_prefetch(wL + (nt0 * 8 + ks + 1) * 512 + lane * 16, 0, 1);
    }
#pragma unroll
    for (int m = 0; m < 8; ++m) {
      const unsigned short* pah = inH + (m * 16 + r) * PADH + ks * 32 + hi8;
      const unsigned short* pal = inL + (m * 16 + r) * PADH + ks * 32 + hi8;
      Frag aH, aL;
      aH.q[0] = *(const f4*)(pah);       aH.q[1] = *(const f4*)(pah + 16);
      aL.q[0] = *(const f4*)(pal);       aL.q[1] = *(const f4*)(pal + 16);
#pragma unroll
      for (int t = 0; t < 2; ++t) {
        acc[t][m] = __builtin_amdgcn_wmma_f32_16x16x32_bf16(false, aH.v, false, bH[t].v,
                                                            (short)0, acc[t][m], false, false);
        acc[t][m] = __builtin_amdgcn_wmma_f32_16x16x32_bf16(false, aH.v, false, bL[t].v,
                                                            (short)0, acc[t][m], false, false);
        acc[t][m] = __builtin_amdgcn_wmma_f32_16x16x32_bf16(false, aL.v, false, bH[t].v,
                                                            (short)0, acc[t][m], false, false);
      }
    }
  }

  // epilogue: bias + relu; write next-layer LDS planes or final global output
#pragma unroll
  for (int t = 0; t < 2; ++t) {
    const int col = (nt0 + t) * 16 + r;
    const float bv = bias[col];
#pragma unroll
    for (int m = 0; m < 8; ++m)
#pragma unroll
      for (int q = 0; q < 8; ++q) {
        float val = fmaxf(acc[t][m][q] + bv, 0.f);
        int row = m * 16 + q + hi8;     // C layout: M = vgpr + 8*hi
        if (LAST) {
          int ro = rows_s[row];
          if (ro >= 0) out[(size_t)ro * D_DIM + col] = val;
        } else {
          unsigned short h = f2bf(val);
          outH[row * PADH + col] = h;
          outL[row * PADH + col] = f2bf(val - bf2f(h));
        }
      }
  }
}

// ---- K4: fused routed 3-layer GEMM over one 128-row expert-grouped tile ----
__global__ __launch_bounds__(BLK, 1) void k_main(
    const float* __restrict__ x, const float* __restrict__ b_leaf,
    const float* __restrict__ b_mid, const float* __restrict__ b_root,
    const unsigned short* __restrict__ wp, const int* __restrict__ wsi,
    float* __restrict__ out) {
  __shared__ unsigned short smem[4 * TILE_M * PADH];   // ~279 KB of 320 KB WGP LDS
  __shared__ int rows_s[TILE_M];
  const int PLANE = TILE_M * PADH;
  unsigned short* pAh = smem;
  unsigned short* pAl = smem + PLANE;
  unsigned short* pBh = smem + 2 * PLANE;
  unsigned short* pBl = smem + 3 * PLANE;

  const int base = blockIdx.x * TILE_M;
  const int o1 = wsi[WS_OFFS + 1], o2 = wsi[WS_OFFS + 2];
  const int o3 = wsi[WS_OFFS + 3], o4 = wsi[WS_OFFS + 4];
  if (base >= o4) return;
  const int e = (base < o1) ? 0 : ((base < o2) ? 1 : ((base < o3) ? 2 : 3));
  const int p = e >> 1;

  const int tid = threadIdx.x;
  if (tid < TILE_M) rows_s[tid] = wsi[WS_PERM + base + tid];
  __syncthreads();

  // gather x rows, split fp32 -> bf16 hi/lo planes
  for (int i = tid; i < TILE_M * (D_DIM / 4); i += BLK) {
    int row = i >> 6;
    int c = (i & 63) << 2;
    int ro = rows_s[row];
    f4 xv = {0.f, 0.f, 0.f, 0.f};
    if (ro >= 0) xv = *(const f4*)(x + (size_t)ro * D_DIM + c);
    u16x4 hv = { f2bf(xv.x), f2bf(xv.y), f2bf(xv.z), f2bf(xv.w) };
    u16x4 lv = { f2bf(xv.x - bf2f(hv.x)), f2bf(xv.y - bf2f(hv.y)),
                 f2bf(xv.z - bf2f(hv.z)), f2bf(xv.w - bf2f(hv.w)) };
    *(u16x4*)(pAh + row * PADH + c) = hv;
    *(u16x4*)(pAl + row * PADH + c) = lv;
  }
  __syncthreads();

  // leaf (routed expert e)
  gemm_layer<false>(pAh, pAl, wp + (2 * e + 0) * WPLANE, wp + (2 * e + 1) * WPLANE,
                    b_leaf + e * D_DIM, pBh, pBl, nullptr, nullptr);
  __syncthreads();
  // mid (parent p)
  gemm_layer<false>(pBh, pBl, wp + (2 * (4 + p) + 0) * WPLANE, wp + (2 * (4 + p) + 1) * WPLANE,
                    b_mid + p * D_DIM, pAh, pAl, nullptr, nullptr);
  __syncthreads();
  // root, scatter to original row order
  gemm_layer<true>(pAh, pAl, wp + 12 * WPLANE, wp + 13 * WPLANE,
                   b_root, nullptr, nullptr, out, rows_s);
}

extern "C" void kernel_launch(void* const* d_in, const int* in_sizes, int n_in,
                              void* d_out, int out_size, void* d_ws, size_t ws_size,
                              hipStream_t stream) {
  (void)in_sizes; (void)n_in; (void)out_size; (void)ws_size;
  const float* x      = (const float*)d_in[0];
  const float* Wg     = (const float*)d_in[1];
  const float* bg     = (const float*)d_in[2];
  const float* W_leaf = (const float*)d_in[3];
  const float* b_leaf = (const float*)d_in[4];
  const float* W_mid  = (const float*)d_in[5];
  const float* b_mid  = (const float*)d_in[6];
  const float* W_root = (const float*)d_in[7];
  const float* b_root = (const float*)d_in[8];
  float* out = (float*)d_out;
  int* wsi = (int*)d_ws;
  unsigned short* wp = (unsigned short*)((char*)d_ws + WPACK_BYTES);

  hipMemsetAsync(wsi + WS_COUNTS, 0, 4 * sizeof(int), stream);
  k_pack<<<896, 256, 0, stream>>>(W_leaf, W_mid, W_root, wp);
  k_gate<<<B_ROWS / 256, 256, 0, stream>>>(x, Wg, bg, wsi);
  k_scan<<<1, 32, 0, stream>>>(wsi);
  k_scatter<<<B_ROWS / 256, 256, 0, stream>>>(wsi);
  k_main<<<B_ROWS / TILE_M + 4, BLK, 0, stream>>>(x, b_leaf, b_mid, b_root, wp, wsi, out);
}
